// Galerkin_Transformer_65326452572863
// MI455X (gfx1250) — matrix operand order repaired and checked by reading, not compiled, not run on hardware
//
#include <hip/hip_runtime.h>
#include <hip/hip_bf16.h>
#include <math.h>
#include <stdint.h>

// ---------------------------------------------------------------------------
// Galerkin Transformer forward for MI455X (gfx1250, wave32, WMMA bf16).
// Large GEMMs: v_wmma_f32_16x16x32_bf16, weight tiles staged into LDS by the
// Tensor Data Mover (double-buffered, TENSORcnt-tracked). Residual fp32.
// ---------------------------------------------------------------------------

#define B_   8
#define N_   4096
#define H_   256
#define NH_  8
#define DH_  32
#define NL_  6
#define DFF_ 1024
#define M_   (B_ * N_)     // 32768 token rows

typedef __bf16 bf16;
typedef __attribute__((ext_vector_type(16))) __bf16 v16bf;
typedef __attribute__((ext_vector_type(8)))  __bf16 bf16x8;
typedef __attribute__((ext_vector_type(8)))  float  v8f;

union FragA { v16bf v; bf16x8 h[2]; };

__device__ __forceinline__ float wsum32(float v) {
#pragma unroll
  for (int o = 16; o > 0; o >>= 1) v += __shfl_xor(v, o, 32);
  return v;
}

__device__ __forceinline__ float gelu_exact(float x) {
  return 0.5f * x * (1.0f + erff(x * 0.70710678118654752f));
}

// ---------------------------------------------------------------------------
// Tensor Data Mover staging of a 64(row) x 32(col) bf16 tile into LDS.
// D# per CDNA5 ISA ch.8: group0 = {count, lds_addr, global_addr, type=2},
// group1 = {data_size=2B, tensor_dim0=K, tensor_dim1=big, tile 32x64,
// stride0=K}. Remaining groups zero (2-D tile). Tracked by TENSORcnt.
// This toolchain exposes the 6-arg builtin:
//   (uint32x4 g0, int32x8 g1, int32x4, int32x4, int32x8, i32 cpol)
// ---------------------------------------------------------------------------
#if defined(__gfx1250__) && __has_builtin(__builtin_amdgcn_tensor_load_to_lds) && \
    __has_builtin(__builtin_amdgcn_s_wait_tensorcnt)
#define USE_TDM 1
#else
#define USE_TDM 0
#endif

#if USE_TDM
typedef unsigned int u32;
typedef u32 u32x4 __attribute__((ext_vector_type(4)));
typedef int  i32x8 __attribute__((ext_vector_type(8)));
typedef int  i32x4 __attribute__((ext_vector_type(4)));

__device__ __forceinline__ void tdm_load_b_tile(const bf16* gsrc, bf16* ldst, int K)
{
  u32 lds_addr = (u32)(uintptr_t)ldst;            // LDS byte address = addr[31:0]
  unsigned long long ga = (unsigned long long)(uintptr_t)gsrc;
  u32x4 g0;
  g0[0] = 1u;                                     // count=1: one valid descriptor
  g0[1] = lds_addr;                               // D#.lds_addr
  g0[2] = (u32)ga;                                // D#.global_addr[31:0]
  g0[3] = (u32)((ga >> 32) & 0x01ffffffu) | (2u << 30);  // addr[56:32], type=2
  u32 td0 = (u32)K;                               // tensor_dim0 (row length)
  i32x8 g1;
  g1[0] = (int)(1u << 16);                        // data_size=1 -> 2-byte elems
  g1[1] = (int)((td0 & 0xffffu) << 16);           // tensor_dim0[15:0] @ bit 48
  g1[2] = (int)(((td0 >> 16) & 0xffffu) |         // tensor_dim0[31:16]
                (0xffffu << 16));                 // tensor_dim1[15:0] = big
  g1[3] = (int)(32u << 16);                       // tensor_dim1[31:16]=0, tile_dim0=32
  g1[4] = (int)64u;                               // tile_dim1=64, tile_dim2=0
  g1[5] = (int)td0;                               // tensor_dim0_stride[31:0] = K
  g1[6] = 0;                                      // stride0[47:32]=0, stride1 lo=0
  g1[7] = 0;                                      // stride1 hi = 0
  i32x4 gz4 = {0, 0, 0, 0};
  i32x8 gz8 = {0, 0, 0, 0, 0, 0, 0, 0};
  __builtin_amdgcn_tensor_load_to_lds(g0, g1, gz4, gz4, gz8, 0);
}
#endif

// ---------------------------------------------------------------------------
// Generic bf16 GEMM: out[M,Nout] = A[M,K] * W[K,Nout], W pre-transposed to
// Wt[Nout,K]. Block = 256 thr = 8 waves; wave -> 16(M) x 64(N) strip; the
// 64x32 Wt tile per k-step is shared block-wide (LDS via TDM when available).
// mode 0: out_bf = acc            (QKV projections, no bias)
// mode 1: out_bf = gelu(acc+bias) (MLP up-projection)
// mode 2: out_f += acc + bias     (residual accumulate into h: Wo, MLP down)
// ---------------------------------------------------------------------------
__global__ __launch_bounds__(256) void gemm_bf16_kernel(
    const bf16* __restrict__ A, const bf16* __restrict__ Wt,
    const float* __restrict__ bias, bf16* __restrict__ outB,
    float* __restrict__ outF, int K, int Nout, int mode)
{
  int tid  = threadIdx.x;
  int wave = tid >> 5, lane = tid & 31;
  int l16  = lane & 15, g = lane >> 4;
  int m0 = blockIdx.y * 128 + wave * 16;
  int n0 = blockIdx.x * 64;

  const bf16* arow = A + (size_t)(m0 + l16) * K;
  v8f acc[4] = {};

#if USE_TDM
  __shared__ __align__(32) bf16 sB[2][64 * 32];   // double-buffered Wt tile, 8KB
  if (wave == 0) {
    tdm_load_b_tile(Wt + (size_t)n0 * K, &sB[0][0], K);
    __builtin_amdgcn_s_wait_tensorcnt(0);
  }
  __syncthreads();
  int nsteps = K >> 5;
  for (int it = 0; it < nsteps; ++it) {
    int k0 = it << 5;
    int cur = it & 1;
    bool more = (it + 1) < nsteps;
    if (more && wave == 0)                        // DMA next tile while computing
      tdm_load_b_tile(Wt + (size_t)n0 * K + (k0 + 32), &sB[cur ^ 1][0], K);
    FragA a;
    a.h[0] = *(const bf16x8*)(arow + k0 + 16 * g);
    a.h[1] = *(const bf16x8*)(arow + k0 + 16 * g + 8);
#pragma unroll
    for (int t = 0; t < 4; ++t) {
      v16bf bv = *(const v16bf*)(&sB[cur][(t * 16 + l16) * 32 + 16 * g]);
      acc[t] = __builtin_amdgcn_wmma_f32_16x16x32_bf16(
          false, a.v, false, bv, (short)0, acc[t], false, false);
    }
    if (more && wave == 0)
      __builtin_amdgcn_s_wait_tensorcnt(0);
    __syncthreads();
  }
#else
  for (int k0 = 0; k0 < K; k0 += 32) {
    FragA a;
    a.h[0] = *(const bf16x8*)(arow + k0 + 16 * g);
    a.h[1] = *(const bf16x8*)(arow + k0 + 16 * g + 8);
#pragma unroll
    for (int t = 0; t < 4; ++t) {
      const bf16* brow = Wt + (size_t)(n0 + t * 16 + l16) * K;
      v16bf bv = *(const v16bf*)(brow + k0 + 16 * g);
      acc[t] = __builtin_amdgcn_wmma_f32_16x16x32_bf16(
          false, a.v, false, bv, (short)0, acc[t], false, false);
    }
  }
#endif

#pragma unroll
  for (int t = 0; t < 4; ++t) {
    int n = n0 + t * 16 + l16;
    float bn = (mode != 0) ? bias[n] : 0.0f;
#pragma unroll
    for (int r = 0; r < 8; ++r) {
      int m = m0 + r + 8 * g;
      float val = acc[t][r];
      if (mode == 0) {
        outB[(size_t)m * Nout + n] = (bf16)val;
      } else if (mode == 1) {
        outB[(size_t)m * Nout + n] = (bf16)gelu_exact(val + bn);
      } else {
        outF[(size_t)m * Nout + n] += val + bn;
      }
    }
  }
}

// ---------------------------------------------------------------------------
// LayerNorm over H=256, fp32 in -> bf16 out. One wave per token row.
// ---------------------------------------------------------------------------
__global__ __launch_bounds__(256) void ln_kernel(
    const float* __restrict__ h, const float* __restrict__ gg,
    const float* __restrict__ bb, bf16* __restrict__ out)
{
  int tid = threadIdx.x, wave = tid >> 5, lane = tid & 31;
  size_t row = (size_t)blockIdx.x * 8 + wave;
  const float4* p4 = (const float4*)(h + row * H_ + lane * 8);
  float4 v0 = p4[0], v1 = p4[1];
  float xv[8] = {v0.x, v0.y, v0.z, v0.w, v1.x, v1.y, v1.z, v1.w};
  float s = 0.f;
#pragma unroll
  for (int i = 0; i < 8; ++i) s += xv[i];
  float mean = wsum32(s) * (1.0f / 256.0f);
  float q = 0.f;
#pragma unroll
  for (int i = 0; i < 8; ++i) { float d = xv[i] - mean; q += d * d; }
  float rs = rsqrtf(wsum32(q) * (1.0f / 256.0f) + 1e-5f);
  bf16* op = out + row * H_ + lane * 8;
#pragma unroll
  for (int i = 0; i < 8; ++i) {
    int c = lane * 8 + i;
    op[i] = (bf16)((xv[i] - mean) * rs * gg[c] + bb[c]);
  }
}

// ---------------------------------------------------------------------------
// Per-head LayerNorm of K and V over DH=32 (in place, bf16). V additionally
// scaled by 1/N (folds the ctx/N normalization). One wave per (b,n,head).
// ---------------------------------------------------------------------------
__global__ __launch_bounds__(256) void kvln_kernel(
    bf16* __restrict__ Kb, bf16* __restrict__ Vb,
    const float* __restrict__ kng, const float* __restrict__ knb,
    const float* __restrict__ vng, const float* __restrict__ vnb)
{
  int tid = threadIdx.x, wave = tid >> 5, lane = tid & 31;
  size_t row = (size_t)blockIdx.x * 8 + wave;   // [0, B*N*NH)
  int hh = (int)(row & 7);
  size_t base = (row >> 3) * H_ + (size_t)hh * DH_;
  int pidx = hh * DH_ + lane;
  {
    float x = (float)Kb[base + lane];
    float mean = wsum32(x) * (1.0f / 32.0f);
    float d = x - mean;
    float var = wsum32(d * d) * (1.0f / 32.0f);
    float rs = rsqrtf(var + 1e-5f);
    Kb[base + lane] = (bf16)(d * rs * kng[pidx] + knb[pidx]);
  }
  {
    float x = (float)Vb[base + lane];
    float mean = wsum32(x) * (1.0f / 32.0f);
    float d = x - mean;
    float var = wsum32(d * d) * (1.0f / 32.0f);
    float rs = rsqrtf(var + 1e-5f);
    Vb[base + lane] = (bf16)((d * rs * vng[pidx] + vnb[pidx]) * (1.0f / (float)N_));
  }
}

__global__ __launch_bounds__(256) void zero_kernel(float* __restrict__ p, int n) {
  int i = blockIdx.x * 256 + threadIdx.x;
  if (i < n) p[i] = 0.0f;
}

// ---------------------------------------------------------------------------
// ctxT[b,h][e][d] = sum_n kn[b,n,h,d] * vn[b,n,h,e]  (vn pre-scaled by 1/N).
// block = (b,h, token-chunk of 512), LDS-tiled 64 tokens, atomic fp32 reduce.
// ---------------------------------------------------------------------------
__global__ __launch_bounds__(256) void ctx_kernel(
    const bf16* __restrict__ Kb, const bf16* __restrict__ Vb,
    float* __restrict__ ctxT)
{
  __shared__ float sk[64 * DH_];
  __shared__ float sv[64 * DH_];
  int bid = blockIdx.x;
  int bh = bid >> 3, chunk = bid & 7;
  int b = bh >> 3, hh = bh & 7;
  int tid = threadIdx.x;
  int d  = tid >> 3;             // output d for this thread's 4 outputs
  int e0 = (tid * 4) & 31;       // first of 4 consecutive e's
  int token = tid >> 2;          // staging: 8 contiguous elems per thread
  int dd0 = (tid & 3) * 8;
  float a0 = 0.f, a1 = 0.f, a2 = 0.f, a3 = 0.f;

  for (int sub = 0; sub < 8; ++sub) {
    int tok0 = chunk * 512 + sub * 64;
    size_t src = ((size_t)(b * N_ + tok0 + token)) * H_ + hh * DH_ + dd0;
    bf16x8 kv = *(const bf16x8*)(Kb + src);
    bf16x8 vv = *(const bf16x8*)(Vb + src);
#pragma unroll
    for (int i = 0; i < 8; ++i) {
      sk[token * DH_ + dd0 + i] = (float)kv[i];
      sv[token * DH_ + dd0 + i] = (float)vv[i];
    }
    __syncthreads();
#pragma unroll 4
    for (int t = 0; t < 64; ++t) {
      float a = sk[t * DH_ + d];
      const float* svp = &sv[t * DH_ + e0];
      a0 += a * svp[0]; a1 += a * svp[1]; a2 += a * svp[2]; a3 += a * svp[3];
    }
    __syncthreads();
  }
  float* dst = ctxT + (size_t)bh * (DH_ * DH_);
  atomicAdd(dst + (e0 + 0) * DH_ + d, a0);
  atomicAdd(dst + (e0 + 1) * DH_ + d, a1);
  atomicAdd(dst + (e0 + 2) * DH_ + d, a2);
  atomicAdd(dst + (e0 + 3) * DH_ + d, a3);
}

// ---------------------------------------------------------------------------
// att[b,n,h*32+e] = sum_d q[b,n,h*32+d] * ctx[d][e]   (single K=32 WMMA step).
// ---------------------------------------------------------------------------
__global__ __launch_bounds__(256) void attq_kernel(
    const bf16* __restrict__ Q, const float* __restrict__ ctxT,
    bf16* __restrict__ att)
{
  int bid = blockIdx.x;          // 64 (b,h) * 32 m-blocks
  int bh = bid >> 5, mblk = bid & 31;
  int b = bh >> 3, hh = bh & 7;
  int tid = threadIdx.x;
  int wave = tid >> 5, lane = tid & 31;
  int l16 = lane & 15, g = lane >> 4;
  int m0 = mblk * 128 + wave * 16;

  const bf16* arow = Q + ((size_t)(b * N_ + m0 + l16)) * H_ + hh * DH_;
  FragA a;
  a.h[0] = *(const bf16x8*)(arow + 16 * g);
  a.h[1] = *(const bf16x8*)(arow + 16 * g + 8);

  v8f acc[2] = {};
#pragma unroll
  for (int t = 0; t < 2; ++t) {
    const float* crow = ctxT + (size_t)bh * (DH_ * DH_) + (size_t)(t * 16 + l16) * DH_;
    v16bf bv;
#pragma unroll
    for (int i = 0; i < 16; ++i) bv[i] = (bf16)crow[16 * g + i];
    acc[t] = __builtin_amdgcn_wmma_f32_16x16x32_bf16(
        false, a.v, false, bv, (short)0, acc[t], false, false);
  }
#pragma unroll
  for (int t = 0; t < 2; ++t)
#pragma unroll
    for (int r = 0; r < 8; ++r) {
      size_t m = (size_t)(b * N_ + m0 + r + 8 * g);
      att[m * H_ + hh * DH_ + t * 16 + l16] = (bf16)acc[t][r];
    }
}

// ---------------------------------------------------------------------------
// Fused bilinear resize (align_corners, channel 2) + unified pos embedding +
// pre-MLP (65 -> 512 gelu -> 256) + placeholder. One block per token.
// ---------------------------------------------------------------------------
__global__ __launch_bounds__(256) void premlp_kernel(
    const float* __restrict__ x, const float* __restrict__ pW1,
    const float* __restrict__ pb1, const float* __restrict__ pW2,
    const float* __restrict__ pb2, const float* __restrict__ ph,
    float* __restrict__ h)
{
  __shared__ float fx[65];
  __shared__ float hid[512];
  int token = blockIdx.x;
  int b = token >> 12;
  int n = token & (N_ - 1);
  int i = n >> 6, j = n & 63;
  int tid = threadIdx.x;
  if (tid == 0) {
    float ci = i * (31.0f / 63.0f);
    int i0 = (int)ci; int i1 = min(i0 + 1, 31); float wi = ci - (float)i0;
    float cj = j * (31.0f / 63.0f);
    int j0 = (int)cj; int j1 = min(j0 + 1, 31); float wj = cj - (float)j0;
    const float* xb = x + (size_t)b * 32 * 32 * 3 + 2;
    float p00 = xb[(i0 * 32 + j0) * 3], p10 = xb[(i1 * 32 + j0) * 3];
    float p01 = xb[(i0 * 32 + j1) * 3], p11 = xb[(i1 * 32 + j1) * 3];
    float r0 = p00 * (1.f - wi) + p10 * wi;
    float r1 = p01 * (1.f - wi) + p11 * wi;
    fx[0] = r0 * (1.f - wj) + r1 * wj;
  }
  if (tid < 64) {
    int ri = tid >> 3, rj = tid & 7;
    float dx = i * (1.0f / 63.0f) - ri * (1.0f / 7.0f);
    float dy = j * (1.0f / 63.0f) - rj * (1.0f / 7.0f);
    fx[1 + tid] = sqrtf(dx * dx + dy * dy);
  }
  __syncthreads();
  for (int o = tid; o < 512; o += 256) {
    float s = pb1[o];
    for (int k = 0; k < 65; ++k) s += fx[k] * pW1[k * 512 + o];
    hid[o] = gelu_exact(s);
  }
  __syncthreads();
  {
    int o = tid;
    float s = pb2[o];
    for (int k = 0; k < 512; ++k) s += hid[k] * pW2[k * 256 + o];
    h[(size_t)token * H_ + o] = s + ph[o];
  }
}

// ---------------------------------------------------------------------------
// Final LN + projection to 1 output per token. One wave per token.
// ---------------------------------------------------------------------------
__global__ __launch_bounds__(256) void out_kernel(
    const float* __restrict__ h, const float* __restrict__ g3,
    const float* __restrict__ b3, const float* __restrict__ oW,
    const float* __restrict__ ob, float* __restrict__ out)
{
  int tid = threadIdx.x, wave = tid >> 5, lane = tid & 31;
  size_t row = (size_t)blockIdx.x * 8 + wave;
  const float4* p4 = (const float4*)(h + row * H_ + lane * 8);
  float4 v0 = p4[0], v1 = p4[1];
  float xv[8] = {v0.x, v0.y, v0.z, v0.w, v1.x, v1.y, v1.z, v1.w};
  float s = 0.f;
#pragma unroll
  for (int i = 0; i < 8; ++i) s += xv[i];
  float mean = wsum32(s) * (1.0f / 256.0f);
  float q = 0.f;
#pragma unroll
  for (int i = 0; i < 8; ++i) { float d = xv[i] - mean; q += d * d; }
  float rs = rsqrtf(wsum32(q) * (1.0f / 256.0f) + 1e-5f);
  float dot = 0.f;
#pragma unroll
  for (int i = 0; i < 8; ++i) {
    int c = lane * 8 + i;
    dot += ((xv[i] - mean) * rs * g3[c] + b3[c]) * oW[c];
  }
  dot = wsum32(dot);
  if (lane == 0) out[row] = dot + ob[0];
}

// ---------------------------------------------------------------------------
// One-shot weight transpose+convert to bf16 Wt[Nout,K] layout, all layers.
// Per-layer element layout: WqT(65536) WkT WvT WoT W1T(262144) W2T(262144).
// ---------------------------------------------------------------------------
__global__ __launch_bounds__(256) void convw_kernel(
    const float* __restrict__ Wq, const float* __restrict__ Wk,
    const float* __restrict__ Wv, const float* __restrict__ Wo,
    const float* __restrict__ W1, const float* __restrict__ W2,
    bf16* __restrict__ dst)
{
  int id = blockIdx.x * 256 + threadIdx.x;
  const int LW = 786432;
  int layer = id / LW;
  int r = id - layer * LW;
  float val;
  if (r < 262144) {
    int seg = r >> 16;
    int r2 = r & 65535;
    int o = r2 >> 8, k = r2 & 255;
    const float* W = (seg == 0) ? Wq : (seg == 1) ? Wk : (seg == 2) ? Wv : Wo;
    val = W[(size_t)layer * 65536 + (size_t)k * 256 + o];
  } else if (r < 524288) {
    int r3 = r - 262144;
    int o = r3 >> 8, k = r3 & 255;          // o in [0,1024)
    val = W1[(size_t)layer * 262144 + (size_t)k * 1024 + o];
  } else {
    int r3 = r - 524288;
    int o = r3 >> 10, k = r3 & 1023;        // o in [0,256)
    val = W2[(size_t)layer * 262144 + (size_t)k * 256 + o];
  }
  dst[id] = (bf16)val;
}

// ---------------------------------------------------------------------------
extern "C" void kernel_launch(void* const* d_in, const int* in_sizes, int n_in,
                              void* d_out, int out_size, void* d_ws, size_t ws_size,
                              hipStream_t stream)
{
  const float* x           = (const float*)d_in[0];
  const float* pre_W1      = (const float*)d_in[1];
  const float* pre_b1      = (const float*)d_in[2];
  const float* pre_W2      = (const float*)d_in[3];
  const float* pre_b2      = (const float*)d_in[4];
  const float* placeholder = (const float*)d_in[5];
  const float* ln1_g  = (const float*)d_in[6];
  const float* ln1_b  = (const float*)d_in[7];
  const float* ln1a_g = (const float*)d_in[8];
  const float* ln1a_b = (const float*)d_in[9];
  const float* Wq = (const float*)d_in[10];
  const float* Wk = (const float*)d_in[11];
  const float* Wv = (const float*)d_in[12];
  const float* Wo = (const float*)d_in[13];
  const float* bo = (const float*)d_in[14];
  const float* kn_g = (const float*)d_in[15];
  const float* kn_b = (const float*)d_in[16];
  const float* vn_g = (const float*)d_in[17];
  const float* vn_b = (const float*)d_in[18];
  const float* ln2_g = (const float*)d_in[19];
  const float* ln2_b = (const float*)d_in[20];
  const float* mlp_W1 = (const float*)d_in[21];
  const float* mlp_b1 = (const float*)d_in[22];
  const float* mlp_W2 = (const float*)d_in[23];
  const float* mlp_b2 = (const float*)d_in[24];
  const float* ln3_g = (const float*)d_in[25];
  const float* ln3_b = (const float*)d_in[26];
  const float* out_W = (const float*)d_in[27];
  const float* out_b = (const float*)d_in[28];

  // Workspace layout (~144 MB). Rm (MLP hidden, 67 MB) aliases Rq/Rk/Rv+spare,
  // which are dead during the MLP phase.
  char* ws = (char*)d_ws;
  float* h    = (float*)(ws + 0);                 // 33.5 MB fp32 residual
  bf16*  Ra   = (bf16*)(ws + 33554432ull);        // xq / attb (bf16, 16.8 MB)
  bf16*  Rb   = (bf16*)(ws + 50331648ull);        // xkv / y
  bf16*  Rq   = (bf16*)(ws + 67108864ull);
  bf16*  Rk   = (bf16*)(ws + 83886080ull);
  bf16*  Rv   = (bf16*)(ws + 100663296ull);
  bf16*  Rm   = (bf16*)(ws + 67108864ull);        // alias: MLP hidden 67 MB
  float* Rctx = (float*)(ws + 134217728ull);      // 256 KB ctxT
  bf16*  Rw   = (bf16*)(ws + 134479872ull);       // 9.4 MB bf16 weights

  convw_kernel<<<18432, 256, 0, stream>>>(Wq, Wk, Wv, Wo, mlp_W1, mlp_W2, Rw);
  premlp_kernel<<<M_, 256, 0, stream>>>(x, pre_W1, pre_b1, pre_W2, pre_b2,
                                        placeholder, h);

  dim3 gN256(4, 256), gN1024(16, 256);
  for (int i = 0; i < NL_; ++i) {
    const bf16* Wb = Rw + (size_t)i * 786432;
    ln_kernel<<<4096, 256, 0, stream>>>(h, ln1_g + i * H_, ln1_b + i * H_, Ra);
    ln_kernel<<<4096, 256, 0, stream>>>(h, ln1a_g + i * H_, ln1a_b + i * H_, Rb);
    gemm_bf16_kernel<<<gN256, 256, 0, stream>>>(Ra, Wb + 0,      nullptr, Rq, nullptr, 256, 256, 0);
    gemm_bf16_kernel<<<gN256, 256, 0, stream>>>(Rb, Wb + 65536,  nullptr, Rk, nullptr, 256, 256, 0);
    gemm_bf16_kernel<<<gN256, 256, 0, stream>>>(Rb, Wb + 131072, nullptr, Rv, nullptr, 256, 256, 0);
    kvln_kernel<<<32768, 256, 0, stream>>>(Rk, Rv, kn_g + i * H_, kn_b + i * H_,
                                           vn_g + i * H_, vn_b + i * H_);
    zero_kernel<<<256, 256, 0, stream>>>(Rctx, 65536);
    ctx_kernel<<<512, 256, 0, stream>>>(Rk, Rv, Rctx);
    attq_kernel<<<2048, 256, 0, stream>>>(Rq, Rctx, Ra);
    gemm_bf16_kernel<<<gN256, 256, 0, stream>>>(Ra, Wb + 196608, bo + i * H_,
                                                nullptr, h, 256, 256, 2);
    ln_kernel<<<4096, 256, 0, stream>>>(h, ln2_g + i * H_, ln2_b + i * H_, Rb);
    gemm_bf16_kernel<<<gN1024, 256, 0, stream>>>(Rb, Wb + 262144, mlp_b1 + i * DFF_,
                                                 Rm, nullptr, 256, 1024, 1);
    gemm_bf16_kernel<<<gN256, 256, 0, stream>>>(Rm, Wb + 524288, mlp_b2 + i * H_,
                                                nullptr, h, 1024, 256, 2);
  }
  out_kernel<<<4096, 256, 0, stream>>>(h, ln3_g, ln3_b, out_W, out_b, (float*)d_out);

  (void)in_sizes; (void)n_in; (void)out_size; (void)ws_size;
}
